// IMVTensorLSTM_35192962023884
// MI455X (gfx1250) — matrix operand order repaired and checked by reading, not compiled, not run on hardware
//
#include <hip/hip_runtime.h>
#include <hip/hip_bf16.h>

typedef __attribute__((ext_vector_type(16))) _Float16 v16h;
typedef __attribute__((ext_vector_type(8)))  _Float16 v8h;
typedef __attribute__((ext_vector_type(8)))  float    v8f;

#define NB 128
#define NTT 256
#define ND 32
#define NN 64

__device__ __forceinline__ float fast_rcp(float v) {
    return __builtin_amdgcn_rcpf(v);   // single v_rcp_f32
}
__device__ __forceinline__ float fast_tanh(float xx) {
#if __has_builtin(__builtin_amdgcn_tanhf)
    return __builtin_amdgcn_tanhf(xx); // single v_tanh_f32 (CDNA5 TRANS op)
#else
    // (e^{2x}-1)/(e^{2x}+1); saturates correctly at +-1
    return 1.0f - 2.0f * fast_rcp(__expf(2.0f * xx) + 1.0f);
#endif
}
__device__ __forceinline__ float fast_sigmoid(float xx) {
    return fast_rcp(1.0f + __expf(-xx));
}

// Phase 1: fused LSTM recurrence + temporal-attention accumulators + per-(b,i) head pre-acts.
// Grid: 64 blocks = (bhalf[2] x dim i[32]); block = 512 threads = 16 waves (wave32).
// Wave (mt, nq): M-tile mt (16 batch rows), N-quarter nq (16 of 64 units), all 4 gates.
__global__ __launch_bounds__(512, 1)
void imv_lstm_phase1(
    const float* __restrict__ x,
    const float* __restrict__ Uj, const float* __restrict__ Ui,
    const float* __restrict__ Uf, const float* __restrict__ Uo,
    const float* __restrict__ Wj, const float* __restrict__ Wi,
    const float* __restrict__ Wf, const float* __restrict__ Wo,
    const float* __restrict__ bj, const float* __restrict__ bi,
    const float* __restrict__ bfv, const float* __restrict__ bo,
    const float* __restrict__ Fa,  const float* __restrict__ Fab,
    const float* __restrict__ Fbw, const float* __restrict__ Fbb,
    const float* __restrict__ Pw,  const float* __restrict__ Pb,
    float* __restrict__ alphas_raw,   // [B][T][D]  (unnormalized)
    float* __restrict__ asum_out,     // [B][D]
    float* __restrict__ mu_out,       // [B][D]
    float* __restrict__ braw_out)     // [B][D]
{
    __shared__ _Float16 hlds[64][72];   // h_t for this block's 64 rows, padded stride (144B = 9 banks)
    __shared__ float    spart[64][4];   // alpha-dot partials per N-quarter
    __shared__ float    hg[64][128];    // final [g_n , h_T] per row

    const int i     = blockIdx.x & 31;
    const int bbase = (blockIdx.x >> 5) * 64;
    const int tid   = threadIdx.x;
    const int wave  = tid >> 5;
    const int lane  = tid & 31;
    const int ln    = lane & 15;
    const int hh    = lane >> 4;        // half-wave select
    const int mt    = wave >> 2;        // 0..3 : 16-row M tile
    const int nq    = wave & 3;         // 0..3 : 16-col N tile
    const int ncol  = nq * 16 + ln;

    for (int k = tid; k < 64 * 72; k += 512) (&hlds[0][0])[k] = (_Float16)0.0f;

    const float* Wp[4] = { Wj, Wi, Wf, Wo };
    const float* Up[4] = { Uj, Ui, Uf, Uo };
    const float* Bp[4] = { bj, bi, bfv, bo };

    // B fragments held in VGPRs for the whole kernel.
    // 16-bit B 32x16 layout: element e -> (K = kt*32 + (e<8 ? hh*8+e : 16+hh*8+e-8), N = ncol)
    v16h bfrag[4][2];
    #pragma unroll
    for (int g = 0; g < 4; ++g)
      #pragma unroll
      for (int kt = 0; kt < 2; ++kt)
        #pragma unroll
        for (int e = 0; e < 16; ++e) {
          const int kk = kt * 32 + (e < 8 ? hh * 8 + e : 16 + hh * 8 + (e - 8));
          bfrag[g][kt][e] = (_Float16)Wp[g][(kk * ND + i) * NN + ncol];
        }

    float ug[4], bg[4];
    #pragma unroll
    for (int g = 0; g < 4; ++g) {
      ug[g] = Up[g][i * NN + ncol];
      bg[g] = Bp[g][i * NN + ncol];
    }
    const float fa  = Fa[ncol * ND + i];   // F_alpha_n[n][i][0]
    const float fab = Fab[i];

    float cst[8], gnum[8], asum[8], hcur[8];
    #pragma unroll
    for (int r = 0; r < 8; ++r) { cst[r] = 0.f; gnum[r] = 0.f; asum[r] = 0.f; hcur[r] = 0.f; }

    __syncthreads();

    const _Float16* hrow = &hlds[mt * 16 + ln][0];

    for (int t = 0; t < NTT; ++t) {
      // A fragments (16-bit A 16x32 layout): lane holds row M=ln;
      // elems 0..7 = K(hh*8..+7), elems 8..15 = K(16+hh*8..+7); a1 is K+32.
      v8h p0 = *(const v8h*)(hrow + hh * 8);
      v8h p1 = *(const v8h*)(hrow + 16 + hh * 8);
      v8h p2 = *(const v8h*)(hrow + 32 + hh * 8);
      v8h p3 = *(const v8h*)(hrow + 48 + hh * 8);
      v16h a0 = __builtin_shufflevector(p0, p1, 0,1,2,3,4,5,6,7,8,9,10,11,12,13,14,15);
      v16h a1 = __builtin_shufflevector(p2, p3, 0,1,2,3,4,5,6,7,8,9,10,11,12,13,14,15);

      v8f acc[4];
      #pragma unroll
      for (int g = 0; g < 4; ++g) {
        v8f z = {0.f,0.f,0.f,0.f,0.f,0.f,0.f,0.f};
        z = __builtin_amdgcn_wmma_f32_16x16x32_f16(false, a0, false, bfrag[g][0], (short)0, z, false, false);
        z = __builtin_amdgcn_wmma_f32_16x16x32_f16(false, a1, false, bfrag[g][1], (short)0, z, false, false);
        acc[g] = z;
      }

      __syncthreads();   // all reads of h_t complete before h_{t+1} stores

      // C layout: element r -> (M = r + hh*8, N = ncol)
      #pragma unroll
      for (int r = 0; r < 8; ++r) {
        const int lrow = mt * 16 + r + hh * 8;
        const float xv = x[((size_t)(bbase + lrow) * NTT + t) * ND + i];
        const float pj = acc[0][r] + xv * ug[0] + bg[0];
        const float pi = acc[1][r] + xv * ug[1] + bg[1];
        const float pf = acc[2][r] + xv * ug[2] + bg[2];
        const float po = acc[3][r] + xv * ug[3] + bg[3];
        const float jg = fast_tanh(pj);
        const float ig = fast_sigmoid(pi);
        const float fg = fast_sigmoid(pf);
        const float og = fast_sigmoid(po);
        cst[r] = cst[r] * fg + ig * jg;
        const float h = og * fast_tanh(cst[r]);
        hcur[r] = h;
        hlds[lrow][ncol] = (_Float16)h;
        // alpha-dot partial over this wave's 16 columns (reduce within 16-lane half)
        float p = h * fa;
        p += __shfl_xor(p, 1, 32);
        p += __shfl_xor(p, 2, 32);
        p += __shfl_xor(p, 4, 32);
        p += __shfl_xor(p, 8, 32);
        if (ln == 0) spart[lrow][nq] = p;
      }

      __syncthreads();   // h_{t+1} + partials visible

      #pragma unroll
      for (int r = 0; r < 8; ++r) {
        const int lrow = mt * 16 + r + hh * 8;
        const float s = spart[lrow][0] + spart[lrow][1] + spart[lrow][2] + spart[lrow][3];
        const float a = __expf(fast_tanh(s + fab));
        asum[r] += a;
        gnum[r] += a * hcur[r];
        if (ln == 0 && nq == 0)
          alphas_raw[((size_t)(bbase + lrow) * NTT + t) * ND + i] = a;
      }
    }

    // Stage g_n = gnum/asum and h_T; emit asum
    #pragma unroll
    for (int r = 0; r < 8; ++r) {
      const int lrow = mt * 16 + r + hh * 8;
      if (ln == 0 && nq == 0)
        asum_out[(bbase + lrow) * ND + i] = asum[r];
      hg[lrow][ncol]      = gnum[r] * fast_rcp(asum[r]);
      hg[lrow][64 + ncol] = hcur[r];
    }
    __syncthreads();

    // Per-row head: mu = Phi_w . [g,hT] + Phi_b ; braw = exp(tanh(F_beta_w . [g,hT] + F_beta_b))
    if (tid < 64) {
      float mu = Pb[0];
      float sb = Fbb[0];
      #pragma unroll 8
      for (int k2 = 0; k2 < 128; ++k2) {
        const float v = hg[tid][k2];
        mu += v * Pw[k2];
        sb += v * Fbw[k2];
      }
      mu_out[(bbase + tid) * ND + i]   = mu;
      braw_out[(bbase + tid) * ND + i] = __expf(fast_tanh(sb));
    }
}

// Phase 2: beta softmax over D and the final mean. One thread per batch row.
__global__ void imv_head(const float* __restrict__ mu, const float* __restrict__ braw,
                         float* __restrict__ mean_out, float* __restrict__ betas_out)
{
    const int b = threadIdx.x;  // 128
    float bs = 0.f;
    #pragma unroll 8
    for (int i = 0; i < ND; ++i) bs += braw[b * ND + i];
    const float inv = fast_rcp(bs);
    float m = 0.f;
    #pragma unroll 8
    for (int i = 0; i < ND; ++i) {
      const float bt = braw[b * ND + i] * inv;
      betas_out[b * ND + i] = bt;
      m += bt * mu[b * ND + i];
    }
    mean_out[b] = m;
}

// Phase 3: normalize alphas in place: alphas[b,t,i] *= 1/asum[b,i]
__global__ void imv_alpha_norm(float* __restrict__ alphas, const float* __restrict__ asum)
{
    const size_t idx = (size_t)blockIdx.x * blockDim.x + threadIdx.x;
    const int i = (int)(idx & 31);
    const int b = (int)(idx >> 13);   // idx / (T*D) = idx / 8192
    alphas[idx] = alphas[idx] * fast_rcp(asum[b * ND + i]);
}

extern "C" void kernel_launch(void* const* d_in, const int* in_sizes, int n_in,
                              void* d_out, int out_size, void* d_ws, size_t ws_size,
                              hipStream_t stream) {
    (void)in_sizes; (void)n_in; (void)out_size; (void)ws_size;
    const float* x   = (const float*)d_in[0];
    const float* Uj  = (const float*)d_in[1];
    const float* Ui  = (const float*)d_in[2];
    const float* Uf  = (const float*)d_in[3];
    const float* Uo  = (const float*)d_in[4];
    const float* Wj  = (const float*)d_in[5];
    const float* Wi  = (const float*)d_in[6];
    const float* Wf  = (const float*)d_in[7];
    const float* Wo  = (const float*)d_in[8];
    const float* bj  = (const float*)d_in[9];
    const float* bi  = (const float*)d_in[10];
    const float* bfv = (const float*)d_in[11];
    const float* bo  = (const float*)d_in[12];
    const float* Fa  = (const float*)d_in[13];
    const float* Fab = (const float*)d_in[14];
    const float* Fbw = (const float*)d_in[15];
    const float* Fbb = (const float*)d_in[16];
    const float* Pw  = (const float*)d_in[17];
    const float* Pb  = (const float*)d_in[18];

    float* out      = (float*)d_out;
    float* mean_out = out;                                   // [B,1]          = 128
    float* alphas   = out + 128;                             // [B,T,D,1]      = 1048576
    float* betas    = out + 128 + (size_t)NB * NTT * ND;     // [B,D,1]        = 4096

    float* ws   = (float*)d_ws;
    float* asum = ws;            // [B][D] = 4096 floats
    float* mu   = ws + 4096;     // [B][D]
    float* braw = ws + 8192;     // [B][D]

    imv_lstm_phase1<<<dim3(64), dim3(512), 0, stream>>>(
        x, Uj, Ui, Uf, Uo, Wj, Wi, Wf, Wo, bj, bi, bfv, bo,
        Fa, Fab, Fbw, Fbb, Pw, Pb,
        alphas, asum, mu, braw);

    imv_head<<<dim3(1), dim3(128), 0, stream>>>(mu, braw, mean_out, betas);

    imv_alpha_norm<<<dim3(4096), dim3(256), 0, stream>>>(alphas, asum);
}
